// Scorer_11287174054654
// MI455X (gfx1250) — compile-verified, matrix operand order checked by reading, not run on hardware
//
#include <hip/hip_runtime.h>
#include <hip/hip_bf16.h>

typedef __attribute__((ext_vector_type(8)))  __bf16 v8bf;
typedef __attribute__((ext_vector_type(16))) __bf16 v16bf;
typedef __attribute__((ext_vector_type(8)))  float  v8f;

#define C_DIM  512
#define N_MB   16384
#define M_FV   8192
#define NIMG   8
#define PIX    1024
#define MTILE  256            // rows per block (8 waves x 32 rows)
#define NCHUNK 2048           // cols per block
#define NGRP   64             // B-panel width (cols) staged in LDS
#define NGROUPS (NCHUNK / NGRP)
#define BUFSZ  131072         // one B panel: 64 rows x 512 x 2B x (hi+lo)
#define HALF_PANEL 65536      // hi region size inside a panel

// ---- CDNA5 async global->LDS staging (guarded; falls back to ds_store) ----
#if defined(__has_builtin)
#if __has_builtin(__builtin_amdgcn_global_load_async_to_lds_b128) && \
    __has_builtin(__builtin_amdgcn_s_wait_asynccnt)
#define HAVE_ASYNC_LDS 1
#endif
#endif
#ifndef HAVE_ASYNC_LDS
#define HAVE_ASYNC_LDS 0
#endif

typedef int v4i __attribute__((__vector_size__(4 * sizeof(int))));
#if HAVE_ASYNC_LDS
typedef __attribute__((address_space(1))) v4i as1_v4i;  // global (printed __device__)
typedef __attribute__((address_space(3))) v4i as3_v4i;  // LDS
#endif

__device__ __forceinline__ void stage_b128(const void* g, void* l) {
#if HAVE_ASYNC_LDS
  // AS1 value == generic VA; AS3 value == low 32 bits of generic LDS VA.
  __builtin_amdgcn_global_load_async_to_lds_b128(
      (as1_v4i*)(unsigned long long)g,
      (as3_v4i*)(unsigned int)(unsigned long long)l, 0, 0);
#else
  *(v8bf*)l = *(const v8bf*)g;
#endif
}

__device__ __forceinline__ void wait_async_all() {
#if HAVE_ASYNC_LDS
  __builtin_amdgcn_s_wait_asynccnt(0);
#endif
}

// Stage one 64-col B panel (full K, hi+lo) into LDS, swizzled as 16B blocks
// indexed [j][r] (j = 8-elem K-block 0..63, r = panel row 0..63).
__device__ __forceinline__ void stage_group(char* dst,
                                            const __hip_bfloat16* __restrict__ mb_hi,
                                            const __hip_bfloat16* __restrict__ mb_lo,
                                            int n0, int tid) {
#pragma unroll
  for (int s = 0; s < 16; ++s) {
    const int id = tid + 256 * s;        // 0..4095
    const int r = id >> 6;               // panel row (coalesced along j)
    const int j = id & 63;               // K-block
    const size_t goff = (size_t)(n0 + r) * C_DIM + j * 8;
    const unsigned loff = (unsigned)(j * 64 + r) * 16u;
    stage_b128(mb_hi + goff, dst + loff);
    stage_b128(mb_lo + goff, dst + HALF_PANEL + loff);
  }
}

__device__ __forceinline__ v16bf lda16(const __hip_bfloat16* p) {
  v8bf x0 = *(const v8bf*)p;
  v8bf x1 = *(const v8bf*)(p + 16);
  return __builtin_shufflevector(x0, x1, 0,1,2,3,4,5,6,7,8,9,10,11,12,13,14,15);
}

// ---------------------------------------------------------------- init minsq
__global__ __launch_bounds__(256) void k_init_min(unsigned int* minsq) {
  int i = blockIdx.x * blockDim.x + threadIdx.x;
  if (i < M_FV) minsq[i] = 0x7F800000u;  // +inf bits
}

// ------------------------------------------- fp32 -> (bf16 hi, bf16 lo) + norms
__global__ __launch_bounds__(256) void k_prep(const float* __restrict__ fv,
                                              const float* __restrict__ mbp,
                                              __hip_bfloat16* __restrict__ fv_hi,
                                              __hip_bfloat16* __restrict__ fv_lo,
                                              __hip_bfloat16* __restrict__ mb_hi,
                                              __hip_bfloat16* __restrict__ mb_lo,
                                              float* __restrict__ fnorm,
                                              float* __restrict__ mnorm) {
  const int row = blockIdx.x;
  const float* src;
  __hip_bfloat16 *dh, *dl;
  float* nout;
  if (row < M_FV) {
    src = fv + (size_t)row * C_DIM;
    dh = fv_hi + (size_t)row * C_DIM;
    dl = fv_lo + (size_t)row * C_DIM;
    nout = fnorm + row;
  } else {
    const int r = row - M_FV;
    src = mbp + (size_t)r * C_DIM;
    dh = mb_hi + (size_t)r * C_DIM;
    dl = mb_lo + (size_t)r * C_DIM;
    nout = mnorm + r;
  }
  float part = 0.f;
  for (int c = threadIdx.x; c < C_DIM; c += 256) {
    float x = src[c];
    part = fmaf(x, x, part);
    __hip_bfloat16 h = __float2bfloat16(x);
    dh[c] = h;
    dl[c] = __float2bfloat16(x - __bfloat162float(h));
  }
  __shared__ float s[256];
  s[threadIdx.x] = part;
  __syncthreads();
  for (int st = 128; st > 0; st >>= 1) {
    if (threadIdx.x < st) s[threadIdx.x] += s[threadIdx.x + st];
    __syncthreads();
  }
  if (threadIdx.x == 0) *nout = s[0];
}

// ---- fused GEMM: bf16 hi/lo x3 WMMA, B panels async-staged in LDS (2x buf) ----
__global__ __launch_bounds__(256) void k_min_gemm(
    const __hip_bfloat16* __restrict__ fv_hi,
    const __hip_bfloat16* __restrict__ fv_lo,
    const __hip_bfloat16* __restrict__ mb_hi,
    const __hip_bfloat16* __restrict__ mb_lo,
    const float* __restrict__ fnorm,
    const float* __restrict__ mnorm,
    unsigned int* __restrict__ minsq) {
  extern __shared__ char smem[];  // 2 x BUFSZ

  const int tid   = threadIdx.x;
  const int lane  = tid & 31;
  const int wave  = tid >> 5;
  const int l15   = lane & 15;
  const int khalf = lane >> 4;
  const int mt0   = blockIdx.x * MTILE + wave * 32;  // 32 rows per wave
  const int nbase = blockIdx.y * NCHUNK;

  const size_t a0base = (size_t)(mt0 + l15) * C_DIM;
  const size_t a1base = a0base + (size_t)16 * C_DIM;

  float fn[2][8];
#pragma unroll
  for (int m = 0; m < 2; ++m)
#pragma unroll
    for (int v = 0; v < 8; ++v) fn[m][v] = fnorm[mt0 + 16 * m + v + 8 * khalf];

  float rmin[2][8];
#pragma unroll
  for (int m = 0; m < 2; ++m)
#pragma unroll
    for (int v = 0; v < 8; ++v) rmin[m][v] = __builtin_inff();

  // prologue: stage panel 0
  stage_group(smem, mb_hi, mb_lo, nbase, tid);
  wait_async_all();
  __syncthreads();

  int gbuf = 0;
  for (int g = 0; g < NGROUPS; ++g) {
    char* bcur = smem + (size_t)gbuf * BUFSZ;
    if (g + 1 < NGROUPS)  // prefetch next panel into other buffer (overlapped)
      stage_group(smem + (size_t)(1 - gbuf) * BUFSZ, mb_hi, mb_lo,
                  nbase + (g + 1) * NGRP, tid);

    const int ng0 = nbase + g * NGRP;
    float mn[4];
#pragma unroll
    for (int t = 0; t < 4; ++t) mn[t] = mnorm[ng0 + t * 16 + l15];

    v8f acc[2][4];
#pragma unroll
    for (int m = 0; m < 2; ++m)
#pragma unroll
      for (int t = 0; t < 4; ++t) acc[m][t] = (v8f){0.f,0.f,0.f,0.f,0.f,0.f,0.f,0.f};

#pragma unroll 1
    for (int kc = 0; kc < C_DIM / 32; ++kc) {
      const int ka = kc * 32 + khalf * 8;
      v16bf A0h = lda16(fv_hi + a0base + ka);
      v16bf A0l = lda16(fv_lo + a0base + ka);
      v16bf A1h = lda16(fv_hi + a1base + ka);
      v16bf A1l = lda16(fv_lo + a1base + ka);
      const unsigned jb = (unsigned)(kc * 4 + khalf);
#pragma unroll
      for (int t = 0; t < 4; ++t) {
        const unsigned off0 = (jb * 64u + (unsigned)(t * 16 + l15)) * 16u;
        v8bf bh0 = *(const v8bf*)(bcur + off0);
        v8bf bh1 = *(const v8bf*)(bcur + off0 + 2048);              // j+2 block
        v8bf bl0 = *(const v8bf*)(bcur + HALF_PANEL + off0);
        v8bf bl1 = *(const v8bf*)(bcur + HALF_PANEL + off0 + 2048);
        v16bf Bh = __builtin_shufflevector(bh0, bh1, 0,1,2,3,4,5,6,7,8,9,10,11,12,13,14,15);
        v16bf Bl = __builtin_shufflevector(bl0, bl1, 0,1,2,3,4,5,6,7,8,9,10,11,12,13,14,15);
        // dot ~= ah*bh + ah*bl + al*bh (fp32 accumulate)
        acc[0][t] = __builtin_amdgcn_wmma_f32_16x16x32_bf16(false, A0h, false, Bh, (short)0, acc[0][t], false, false);
        acc[0][t] = __builtin_amdgcn_wmma_f32_16x16x32_bf16(false, A0h, false, Bl, (short)0, acc[0][t], false, false);
        acc[0][t] = __builtin_amdgcn_wmma_f32_16x16x32_bf16(false, A0l, false, Bh, (short)0, acc[0][t], false, false);
        acc[1][t] = __builtin_amdgcn_wmma_f32_16x16x32_bf16(false, A1h, false, Bh, (short)0, acc[1][t], false, false);
        acc[1][t] = __builtin_amdgcn_wmma_f32_16x16x32_bf16(false, A1h, false, Bl, (short)0, acc[1][t], false, false);
        acc[1][t] = __builtin_amdgcn_wmma_f32_16x16x32_bf16(false, A1l, false, Bh, (short)0, acc[1][t], false, false);
      }
    }

#pragma unroll
    for (int m = 0; m < 2; ++m)
#pragma unroll
      for (int t = 0; t < 4; ++t)
#pragma unroll
        for (int v = 0; v < 8; ++v) {
          float sq = fmaxf(fn[m][v] + mn[t] - 2.0f * acc[m][t][v], 0.0f);
          rmin[m][v] = fminf(rmin[m][v], sq);
        }

    wait_async_all();   // my stage of next panel complete
    __syncthreads();    // everyone's stage complete, everyone done reading bcur
    gbuf ^= 1;
  }

  // per-row min over the 16 column-lanes of each half, then atomicMin per row.
#pragma unroll
  for (int m = 0; m < 2; ++m)
#pragma unroll
    for (int v = 0; v < 8; ++v) {
      float x = rmin[m][v];
#pragma unroll
      for (int mask = 8; mask > 0; mask >>= 1)
        x = fminf(x, __shfl_xor(x, mask, 16));
      if (l15 == 0)  // non-negative floats: uint compare == float compare
        atomicMin(&minsq[mt0 + 16 * m + v + 8 * khalf], __float_as_uint(x));
    }
}

// ---------------------------- sqrt -> pixel scores; first-index argmax / image
__global__ __launch_bounds__(256) void k_pixel_argmax(
    const unsigned int* __restrict__ minsq, float* __restrict__ out_pixel,
    int* __restrict__ winner) {
  const int img = blockIdx.x, t = threadIdx.x;
  float best = -1.f;
  int bidx = 0;
  for (int p = t; p < PIX; p += 256) {
    const int i = img * PIX + p;
    float s = sqrtf(__uint_as_float(minsq[i]));
    out_pixel[i] = s;
    if (s > best || (s == best && p < bidx)) { best = s; bidx = p; }
  }
  __shared__ float sv[256];
  __shared__ int si[256];
  sv[t] = best; si[t] = bidx;
  __syncthreads();
  for (int st = 128; st > 0; st >>= 1) {
    if (t < st) {
      if (sv[t + st] > sv[t] || (sv[t + st] == sv[t] && si[t + st] < si[t])) {
        sv[t] = sv[t + st]; si[t] = si[t + st];
      }
    }
    __syncthreads();
  }
  if (t == 0) winner[img] = si[0];
}

// ------------- exact fp32 top-b at the winner pixel of each image + softmax
__global__ __launch_bounds__(1024) void k_topk_image(
    const float* __restrict__ fv, const float* __restrict__ mbp,
    const int* __restrict__ winner, const int* __restrict__ bptr,
    float* __restrict__ out_img) {
  extern __shared__ float dist[];  // N_MB floats (64KB dynamic LDS)
  __shared__ float frow[C_DIM];
  __shared__ float wval[32];
  __shared__ int   widx[32];
  __shared__ float s_top[32];

  const int img = blockIdx.x;
  const int t = threadIdx.x;
  const int B = *bptr;
  const int nb = (B < 32) ? B : 32;
  const int row = img * PIX + winner[img];

  for (int c = t; c < C_DIM; c += 1024) frow[c] = fv[(size_t)row * C_DIM + c];
  __syncthreads();

  for (int j = t; j < N_MB; j += 1024) {
    const float* mrow = mbp + (size_t)j * C_DIM;
    float acc = 0.f;
#pragma unroll 4
    for (int k = 0; k < C_DIM; ++k) {
      float d = frow[k] - mrow[k];
      acc = fmaf(d, d, acc);
    }
    dist[j] = acc;
  }
  __syncthreads();

  for (int it = 0; it < nb; ++it) {
    float v = __builtin_inff();
    int vi = N_MB;
    for (int j = t; j < N_MB; j += 1024) {
      float d = dist[j];
      if (d < v || (d == v && j < vi)) { v = d; vi = j; }
    }
#pragma unroll
    for (int mask = 16; mask > 0; mask >>= 1) {
      float ov = __shfl_xor(v, mask, 32);
      int   oi = __shfl_xor(vi, mask, 32);
      if (ov < v || (ov == v && oi < vi)) { v = ov; vi = oi; }
    }
    if ((t & 31) == 0) { wval[t >> 5] = v; widx[t >> 5] = vi; }
    __syncthreads();
    if (t == 0) {
      float bv = wval[0];
      int bi = widx[0];
      for (int w = 1; w < 32; ++w)
        if (wval[w] < bv || (wval[w] == bv && widx[w] < bi)) { bv = wval[w]; bi = widx[w]; }
      s_top[it] = bv;
      if (bi < N_MB) dist[bi] = __builtin_inff();
    }
    __syncthreads();
  }

  if (t == 0) {
    float s[32];
    for (int i = 0; i < nb; ++i) s[i] = sqrtf(fmaxf(s_top[i], 0.f));
    float score = s[0];
    if (nb > 1) {
      float mx = s[0];
      for (int i = 1; i < nb; ++i) mx = fmaxf(mx, s[i]);
      float sum = 0.f;
      for (int i = 0; i < nb; ++i) sum += expf(s[i] - mx);
      score = s[0] * (1.f - expf(s[0] - mx) / sum);
    }
    out_img[img] = score;
  }
}

extern "C" void kernel_launch(void* const* d_in, const int* in_sizes, int n_in,
                              void* d_out, int out_size, void* d_ws, size_t ws_size,
                              hipStream_t stream) {
  (void)in_sizes; (void)n_in; (void)out_size; (void)ws_size;
  const float* fv  = (const float*)d_in[0];  // (8,32,32,512) -> (8192,512)
  const float* mbp = (const float*)d_in[1];  // (16384,512)
  const int* bptr  = (const int*)d_in[2];    // b = 9

  // workspace carve-up (~50.5 MB)
  char* w = (char*)d_ws;
  __hip_bfloat16* fv_hi = (__hip_bfloat16*)w; w += (size_t)M_FV * C_DIM * 2;
  __hip_bfloat16* fv_lo = (__hip_bfloat16*)w; w += (size_t)M_FV * C_DIM * 2;
  __hip_bfloat16* mb_hi = (__hip_bfloat16*)w; w += (size_t)N_MB * C_DIM * 2;
  __hip_bfloat16* mb_lo = (__hip_bfloat16*)w; w += (size_t)N_MB * C_DIM * 2;
  float* fnorm = (float*)w;        w += (size_t)M_FV * 4;
  float* mnorm = (float*)w;        w += (size_t)N_MB * 4;
  unsigned int* minsq = (unsigned int*)w; w += (size_t)M_FV * 4;
  int* winner = (int*)w;           w += 64;

  float* out_pixel = (float*)d_out;       // 8192 floats (8,1,32,32)
  float* out_img   = out_pixel + M_FV;    // 8 floats

  k_init_min<<<(M_FV + 255) / 256, 256, 0, stream>>>(minsq);
  k_prep<<<M_FV + N_MB, 256, 0, stream>>>(fv, mbp, fv_hi, fv_lo, mb_hi, mb_lo,
                                          fnorm, mnorm);
  dim3 g2(M_FV / MTILE, N_MB / NCHUNK);  // 32 x 8 workgroups
  k_min_gemm<<<g2, 256, 2 * BUFSZ, stream>>>(fv_hi, fv_lo, mb_hi, mb_lo, fnorm,
                                             mnorm, minsq);
  k_pixel_argmax<<<NIMG, 256, 0, stream>>>(minsq, out_pixel, winner);
  k_topk_image<<<NIMG, 1024, N_MB * sizeof(float), stream>>>(fv, mbp, winner,
                                                             bptr, out_img);
}
